// TransformerDecoder_22969485099119
// MI455X (gfx1250) — compile-verified, matrix operand order checked by reading, not run on hardware
//
#include <hip/hip_runtime.h>
#include <math.h>

typedef _Float16 v16h __attribute__((ext_vector_type(16)));
typedef _Float16 v8h  __attribute__((ext_vector_type(8)));
typedef float    v8f  __attribute__((ext_vector_type(8)));

#define D_MODEL 256
#define NQ      900
#define BS      4
#define NTEXT   256
#define HWTOT   13294
#define MROWS   (BS * NQ)          // 3600
#define HS_OUT  ((size_t)6 * BS * NQ * D_MODEL)   // 5,529,600 floats

// ---------------------------------------------------------------------------
// Fragment loader: row-major f16 matrix -> WMMA fragment (two b128 loads).
// A-matrix 16x32 f16 layout (ISA 7.12.2): lane L holds row m0+(L&15),
// K = koff+{0..7} in v[0..7] and koff+16+{0..7} in v[8..15], koff=(L>=16)*8.
// B fragment for B = W^T uses the identical mapping applied to W (N x K).
// ---------------------------------------------------------------------------
__device__ __forceinline__ void load_frag_h(const _Float16* __restrict__ base,
                                            int row0, int maxrow, int ld,
                                            int k0, int lane, v16h& f) {
  int r = row0 + (lane & 15);
  if (r > maxrow) r = maxrow;                 // clamp: garbage rows never stored
  const _Float16* p = base + (size_t)r * ld + k0 + ((lane >> 4) << 3);
  v8h lo = *(const v8h*)(p);                  // 16 B
  v8h hi = *(const v8h*)(p + 16);             // 16 B
  #pragma unroll
  for (int i = 0; i < 8; ++i) { f[i] = lo[i]; f[8 + i] = hi[i]; }
}

// C = act(A[M,K] @ W[N,K]^T + bias[N]); 8 waves/block, each wave 16x64 tile.
// f16out selects f16 or f32 destination.
__global__ void __launch_bounds__(256) gemm_wmma_kernel(
    const _Float16* __restrict__ A, const _Float16* __restrict__ W,
    const float* __restrict__ bias, void* __restrict__ C,
    int M, int N, int K, int relu, int f16out)
{
  int lane = threadIdx.x & 31;
  int wave = threadIdx.x >> 5;
  int m0 = blockIdx.y * 64  + (wave & 3) * 16;
  int n0 = blockIdx.x * 128 + (wave >> 2) * 64;
  if (m0 >= M || n0 >= N) return;

  v8f acc[4];
  #pragma unroll
  for (int j = 0; j < 4; ++j)
    #pragma unroll
    for (int e = 0; e < 8; ++e) acc[j][e] = 0.0f;

  for (int k0 = 0; k0 < K; k0 += 32) {
    __builtin_prefetch(A + (size_t)m0 * K + k0 + 32, 0, 0);  // global_prefetch_b8
    v16h a;
    load_frag_h(A, m0, M - 1, K, k0, lane, a);
    #pragma unroll
    for (int j = 0; j < 4; ++j) {
      v16h b;
      load_frag_h(W, n0 + 16 * j, N - 1, K, k0, lane, b);
      acc[j] = __builtin_amdgcn_wmma_f32_16x16x32_f16(
          false, a, false, b, (short)0, acc[j], false, false);
    }
  }
  // C/D layout: element e -> row m0 + e + (lane>=16 ? 8 : 0), col = lane&15.
  int cb   = lane & 15;
  int radd = (lane >> 4) * 8;
  #pragma unroll
  for (int j = 0; j < 4; ++j) {
    int col = n0 + 16 * j + cb;
    if (col >= N) continue;
    float bv = bias ? bias[col] : 0.0f;
    #pragma unroll
    for (int e = 0; e < 8; ++e) {
      int r = m0 + radd + e;
      if (r < M) {
        float v = acc[j][e] + bv;
        if (relu) v = fmaxf(v, 0.0f);
        if (f16out) ((_Float16*)C)[(size_t)r * N + col] = (_Float16)v;
        else        ((float*)C)[(size_t)r * N + col]    = v;
      }
    }
  }
}

// fp32 -> f16 bulk convert
__global__ void __launch_bounds__(256) cvt_kernel(
    const float* __restrict__ src, _Float16* __restrict__ dst, int n)
{
  int i = blockIdx.x * 256 + threadIdx.x;
  if (i < n) dst[i] = (_Float16)src[i];
}

// f16 dst = A + B (query-pos add fused with convert)
__global__ void __launch_bounds__(256) add_cvt_kernel(
    const float* __restrict__ A, const float* __restrict__ B,
    _Float16* __restrict__ Y, int n)
{
  int i = blockIdx.x * 256 + threadIdx.x;
  if (i < n) Y[i] = (_Float16)(A[i] + B[i]);
}

// ---------------------------------------------------------------------------
// MHA core: one wave per (b,h,q). Scores in LDS, shuffle reductions, lane=dim.
// Output written directly in f16 (feeds only the out-projection GEMM).
// ---------------------------------------------------------------------------
#define ATT_MAX_LK 928
__global__ void __launch_bounds__(256) attn_kernel(
    const float* __restrict__ Q, const float* __restrict__ Kt,
    const float* __restrict__ V, const unsigned char* __restrict__ mask,
    _Float16* __restrict__ O, int B, int H, int Lq, int Lk, float scale)
{
  __shared__ float sc[8][ATT_MAX_LK];
  int lane = threadIdx.x & 31;
  int wave = threadIdx.x >> 5;
  int gid  = blockIdx.x * 8 + wave;
  if (gid >= B * H * Lq) return;
  int q  = gid % Lq;
  int bh = gid / Lq;
  int h  = bh % H;
  int b  = bh / H;
  int D  = H * 32;

  const float* qp = Q + ((size_t)b * Lq + q) * D + h * 32;
  float qreg[32];
  #pragma unroll
  for (int d = 0; d < 32; ++d) qreg[d] = qp[d];

  float mx = -1e30f;
  for (int k = lane; k < Lk; k += 32) {
    const float* kp = Kt + ((size_t)b * Lk + k) * D + h * 32;
    float s = 0.0f;
    #pragma unroll
    for (int d = 0; d < 32; ++d) s += qreg[d] * kp[d];
    s *= scale;
    if (mask && mask[b * Lk + k]) s = -1e9f;
    sc[wave][k] = s;
    mx = fmaxf(mx, s);
  }
  #pragma unroll
  for (int o = 16; o > 0; o >>= 1) mx = fmaxf(mx, __shfl_xor(mx, o, 32));

  float sum = 0.0f;
  for (int k = lane; k < Lk; k += 32) {
    float e = __expf(sc[wave][k] - mx);
    sc[wave][k] = e;
    sum += e;
  }
  #pragma unroll
  for (int o = 16; o > 0; o >>= 1) sum += __shfl_xor(sum, o, 32);
  float inv = 1.0f / sum;

  float acc = 0.0f;
  for (int k = 0; k < Lk; ++k)
    acc += sc[wave][k] * V[((size_t)b * Lk + k) * D + h * 32 + lane];
  O[((size_t)b * Lq + q) * D + h * 32 + lane] = (_Float16)(acc * inv);
}

// ---------------------------------------------------------------------------
// Residual add + LayerNorm (D=256), one wave per row. T may be null.
// Optionally mirrors the result into an f16 buffer for downstream GEMMs.
// ---------------------------------------------------------------------------
__global__ void __launch_bounds__(256) add_ln_kernel(
    const float* __restrict__ X, const float* __restrict__ T,
    const float* __restrict__ g, const float* __restrict__ bt,
    float* __restrict__ Y, _Float16* __restrict__ Y16, int M)
{
  int lane = threadIdx.x & 31;
  int wave = threadIdx.x >> 5;
  int row  = blockIdx.x * 8 + wave;
  if (row >= M) return;
  const float* xp = X + (size_t)row * 256;
  float v[8];
  #pragma unroll
  for (int i = 0; i < 8; ++i) {
    int d = lane + 32 * i;
    v[i] = xp[d] + (T ? T[(size_t)row * 256 + d] : 0.0f);
  }
  float s = 0.0f;
  #pragma unroll
  for (int i = 0; i < 8; ++i) s += v[i];
  #pragma unroll
  for (int o = 16; o > 0; o >>= 1) s += __shfl_xor(s, o, 32);
  float mean = s * (1.0f / 256.0f);
  float var = 0.0f;
  #pragma unroll
  for (int i = 0; i < 8; ++i) { float dv = v[i] - mean; var += dv * dv; }
  #pragma unroll
  for (int o = 16; o > 0; o >>= 1) var += __shfl_xor(var, o, 32);
  float rs = rsqrtf(var * (1.0f / 256.0f) + 1e-5f);
  #pragma unroll
  for (int i = 0; i < 8; ++i) {
    int d = lane + 32 * i;
    float yv = (v[i] - mean) * rs * g[d] + bt[d];
    Y[(size_t)row * 256 + d] = yv;
    if (Y16) Y16[(size_t)row * 256 + d] = (_Float16)yv;
  }
}

// (S,B,D) -> (B,S,D), optional fp32 and/or f16 destinations
__global__ void __launch_bounds__(256) sbd_to_bsd_kernel(
    const float* __restrict__ src, float* __restrict__ dst,
    _Float16* __restrict__ dst16, int S, int B, int D)
{
  size_t i = (size_t)blockIdx.x * 256 + threadIdx.x;
  size_t total = (size_t)S * B * D;
  if (i >= total) return;
  int d = (int)(i % D);
  size_t sb = i / D;
  int b = (int)(sb % B);
  size_t s = sb / B;
  size_t o = ((size_t)b * S + s) * D + d;
  float v = src[i];
  if (dst)   dst[o] = v;
  if (dst16) dst16[o] = (_Float16)v;
}

// sigmoid(refpoints) -> RS (b,q,4) and refs output slice
__global__ void __launch_bounds__(256) refs_kernel(
    const float* __restrict__ refu, float* __restrict__ RS, float* __restrict__ outrefs)
{
  int i = blockIdx.x * 256 + threadIdx.x;
  if (i >= MROWS * 4) return;
  int c = i % 4; int qb = i / 4; int b = qb % BS; int q = qb / BS;
  float v = 1.0f / (1.0f + __expf(-refu[i]));
  int o = (b * NQ + q) * 4 + c;
  RS[o] = v; outrefs[o] = v;
}

// DINO sine embedding of level-0 reference points: rows (b,q) x 512, f16 out
__global__ void __launch_bounds__(256) qse_kernel(
    const float* __restrict__ RS, const float* __restrict__ vr,
    _Float16* __restrict__ QSE)
{
  int i = blockIdx.x * 256 + threadIdx.x;
  if (i >= MROWS * 512) return;
  int e = i % 512;
  int r = i / 512;
  int b = r / NQ;
  int csel = e >> 7;           // order: [y, x, w, h]
  int j2   = e & 127;
  int j    = j2 >> 1;
  const int comp[4] = {1, 0, 2, 3};
  const int vrc[4]  = {1, 0, 0, 1};
  float c = RS[r * 4 + comp[csel]] * vr[(b * 4 + 0) * 2 + vrc[csel]];
  float ang = c * 6.283185307179586f / __powf(10000.0f, (float)j * (1.0f / 64.0f));
  QSE[i] = (_Float16)((j2 & 1) ? __cosf(ang) : __sinf(ang));
}

// softmax over groups of 16 (attention weights per head), in place
__global__ void __launch_bounds__(256) softmax16_kernel(float* __restrict__ AW)
{
  int t = blockIdx.x * 256 + threadIdx.x;
  if (t >= MROWS * 8) return;
  float* p = AW + (size_t)(t / 8) * 128 + (t % 8) * 16;
  float mx = -1e30f;
  #pragma unroll
  for (int k = 0; k < 16; ++k) mx = fmaxf(mx, p[k]);
  float e[16], s = 0.0f;
  #pragma unroll
  for (int k = 0; k < 16; ++k) { e[k] = __expf(p[k] - mx); s += e[k]; }
  float inv = 1.0f / s;
  #pragma unroll
  for (int k = 0; k < 16; ++k) p[k] = e[k] * inv;
}

// deformable sampling: one wave per (b,q,h); lane = channel within head.
// Values gathered from the f16 projected-value tensor, output f16.
__global__ void __launch_bounds__(256) deform_kernel(
    const float* __restrict__ OFF, const float* __restrict__ AW,
    const float* __restrict__ RS, const float* __restrict__ vr,
    const _Float16* __restrict__ VP, _Float16* __restrict__ AO)
{
  const int Hs[4] = {100, 50, 25, 13};
  const int Ss[4] = {0, 10000, 12500, 13125};
  int lane = threadIdx.x & 31;
  int wave = threadIdx.x >> 5;
  int gid  = blockIdx.x * 8 + wave;
  if (gid >= MROWS * 8) return;
  int h  = gid % 8;
  int bq = gid / 8;
  int b  = bq / NQ;
  float cx0 = RS[bq * 4 + 0], cy0 = RS[bq * 4 + 1];
  float w0  = RS[bq * 4 + 2], h0  = RS[bq * 4 + 3];
  const float* offp = OFF + (size_t)bq * 256 + h * 32;
  const float* awp  = AW  + (size_t)bq * 128 + h * 16;
  float acc = 0.0f;
  #pragma unroll
  for (int l = 0; l < 4; ++l) {
    float vx = vr[(b * 4 + l) * 2 + 0], vy = vr[(b * 4 + l) * 2 + 1];
    float cx = cx0 * vx, cy = cy0 * vy, ww = w0 * vx, hh = h0 * vy;
    int Wl = Hs[l], Hl = Hs[l];
    const _Float16* vbase = VP + ((size_t)b * HWTOT + Ss[l]) * 256 + h * 32 + lane;
    #pragma unroll
    for (int p = 0; p < 4; ++p) {
      float lx = cx + offp[(l * 4 + p) * 2 + 0] * 0.25f * ww * 0.5f;
      float ly = cy + offp[(l * 4 + p) * 2 + 1] * 0.25f * hh * 0.5f;
      float x = lx * (float)Wl - 0.5f, y = ly * (float)Hl - 0.5f;
      float x0 = floorf(x), y0 = floorf(y);
      float wx = x - x0, wy = y - y0;
      int xi = (int)x0, yi = (int)y0;
      float s = 0.0f;
      #pragma unroll
      for (int dy = 0; dy < 2; ++dy)
        #pragma unroll
        for (int dx = 0; dx < 2; ++dx) {
          int ix = xi + dx, iy = yi + dy;
          if (ix >= 0 && ix < Wl && iy >= 0 && iy < Hl) {
            float wgt = (dx ? wx : 1.0f - wx) * (dy ? wy : 1.0f - wy);
            s += wgt * (float)vbase[((size_t)iy * Wl + ix) * 256];
          }
        }
      acc += awp[l * 4 + p] * s;
    }
  }
  AO[(size_t)bq * 256 + h * 32 + lane] = (_Float16)acc;
}

// ---------------------------------------------------------------------------
// Host orchestration
// ---------------------------------------------------------------------------
static inline void gemm(const _Float16* A, const _Float16* W, const float* bias,
                        void* C, int M, int N, int K, int relu, int f16out,
                        hipStream_t s) {
  dim3 g((N + 127) / 128, (M + 63) / 64);
  gemm_wmma_kernel<<<g, 256, 0, s>>>(A, W, bias, C, M, N, K, relu, f16out);
}

static inline void cvt(const float* src, _Float16* dst, int n, hipStream_t s) {
  cvt_kernel<<<(n + 255) / 256, 256, 0, s>>>(src, dst, n);
}

extern "C" void kernel_launch(void* const* d_in, const int* in_sizes, int n_in,
                              void* d_out, int out_size, void* d_ws, size_t ws_size,
                              hipStream_t stream) {
  const float* tgt   = (const float*)d_in[0];
  const float* mem   = (const float*)d_in[1];
  const float* refu  = (const float*)d_in[2];
  const float* vr    = (const float*)d_in[3];
  const float* mt    = (const float*)d_in[4];   // (BS, NTEXT, D) batch-first
  const unsigned char* tam = (const unsigned char*)d_in[5];
  const float* sa_w  = (const float*)d_in[8];
  const float* sa_b  = (const float*)d_in[9];
  const float* sa_ow = (const float*)d_in[10];
  const float* sa_ob = (const float*)d_in[11];
  const float* ct_w  = (const float*)d_in[12];
  const float* ct_b  = (const float*)d_in[13];
  const float* ct_ow = (const float*)d_in[14];
  const float* ct_ob = (const float*)d_in[15];
  const float* off_w = (const float*)d_in[16];
  const float* off_b = (const float*)d_in[17];
  const float* aw_w  = (const float*)d_in[18];
  const float* aw_b  = (const float*)d_in[19];
  const float* vp_w  = (const float*)d_in[20];
  const float* vp_b  = (const float*)d_in[21];
  const float* op_w  = (const float*)d_in[22];
  const float* op_b  = (const float*)d_in[23];
  const float* l1_w  = (const float*)d_in[24];
  const float* l1_b  = (const float*)d_in[25];
  const float* l2_w  = (const float*)d_in[26];
  const float* l2_b  = (const float*)d_in[27];
  const float* n1_g  = (const float*)d_in[28];
  const float* n1_b  = (const float*)d_in[29];
  const float* n2_g  = (const float*)d_in[30];
  const float* n2_b  = (const float*)d_in[31];
  const float* cn_g  = (const float*)d_in[32];
  const float* cn_b  = (const float*)d_in[33];
  const float* n3_g  = (const float*)d_in[34];
  const float* n3_b  = (const float*)d_in[35];
  const float* nm_g  = (const float*)d_in[36];
  const float* nm_b  = (const float*)d_in[37];
  const float* rph_w1 = (const float*)d_in[38];
  const float* rph_b1 = (const float*)d_in[39];
  const float* rph_w2 = (const float*)d_in[40];
  const float* rph_b2 = (const float*)d_in[41];
  float* out = (float*)d_out;

  // workspace carve-up (256-byte aligned chunks)
  char* wsb = (char*)d_ws;
  size_t off = 0;
  auto allocB = [&](size_t bytes) {
    void* p = wsb + off;
    off += (bytes + 255) & ~(size_t)255;
    return p;
  };
  const int nX = MROWS * 256;
  float* X    = (float*)allocB((size_t)nX * 4);
  float* QP   = (float*)allocB((size_t)nX * 4);
  float* Qb   = (float*)allocB((size_t)nX * 4);
  float* Kb   = (float*)allocB((size_t)nX * 4);
  float* Vb   = (float*)allocB((size_t)nX * 4);
  float* T2   = (float*)allocB((size_t)nX * 4);
  float* OFb  = (float*)allocB((size_t)nX * 4);
  float* AWb  = (float*)allocB((size_t)MROWS * 128 * 4);
  float* RS   = (float*)allocB((size_t)MROWS * 4 * 4);
  _Float16* X16   = (_Float16*)allocB((size_t)nX * 2);
  _Float16* XQ16  = (_Float16*)allocB((size_t)nX * 2);
  _Float16* AO16  = (_Float16*)allocB((size_t)nX * 2);
  _Float16* TMP16 = (_Float16*)allocB((size_t)nX * 2);
  _Float16* QSE16 = (_Float16*)allocB((size_t)MROWS * 512 * 2);
  _Float16* MT16  = (_Float16*)allocB((size_t)BS * NTEXT * 256 * 2);
  _Float16* FF16  = (_Float16*)allocB((size_t)MROWS * 2048 * 2);
  _Float16* MB16  = (_Float16*)allocB((size_t)BS * HWTOT * 256 * 2);
  _Float16* VP16  = (_Float16*)allocB((size_t)BS * HWTOT * 256 * 2);
  _Float16* W16   = (_Float16*)allocB((size_t)2048 * 256 * 2);

  const float scale = 0.17677669529663689f;   // 1/sqrt(32)
  const size_t memTot = (size_t)BS * HWTOT * 256;

  // ---- loop-invariant prep ----
  sbd_to_bsd_kernel<<<(nX + 255) / 256, 256, 0, stream>>>(tgt, X, X16, NQ, BS, 256);
  sbd_to_bsd_kernel<<<(int)((memTot + 255) / 256), 256, 0, stream>>>(
      mem, nullptr, MB16, HWTOT, BS, 256);
  cvt(mt, MT16, BS * NTEXT * 256, stream);
  refs_kernel<<<(MROWS * 4 + 255) / 256, 256, 0, stream>>>(refu, RS, out + HS_OUT);
  qse_kernel<<<(MROWS * 512 + 255) / 256, 256, 0, stream>>>(RS, vr, QSE16);
  cvt(rph_w1, W16, 256 * 512, stream);
  gemm(QSE16, W16, rph_b1, TMP16, MROWS, 256, 512, 1, 1, stream);      // relu MLP
  cvt(rph_w2, W16, 256 * 256, stream);
  gemm(TMP16, W16, rph_b2, QP, MROWS, 256, 256, 0, 0, stream);

  for (int i = 0; i < 6; ++i) {
    // ---- self attention ----
    add_cvt_kernel<<<(nX + 255) / 256, 256, 0, stream>>>(X, QP, XQ16, nX);
    cvt(sa_w + (size_t)i * 768 * 256, W16, 768 * 256, stream);
    gemm(XQ16, W16,               sa_b + (size_t)i * 768,       Qb, MROWS, 256, 256, 0, 0, stream);
    gemm(XQ16, W16 + 256 * 256,   sa_b + (size_t)i * 768 + 256, Kb, MROWS, 256, 256, 0, 0, stream);
    gemm(X16,  W16 + 512 * 256,   sa_b + (size_t)i * 768 + 512, Vb, MROWS, 256, 256, 0, 0, stream);
    attn_kernel<<<(BS * 8 * NQ + 7) / 8, 256, 0, stream>>>(
        Qb, Kb, Vb, nullptr, AO16, BS, 8, NQ, NQ, scale);
    cvt(sa_ow + (size_t)i * 65536, W16, 65536, stream);
    gemm(AO16, W16, sa_ob + (size_t)i * 256, T2, MROWS, 256, 256, 0, 0, stream);
    add_ln_kernel<<<(MROWS + 7) / 8, 256, 0, stream>>>(
        X, T2, n2_g + (size_t)i * 256, n2_b + (size_t)i * 256, X, nullptr, MROWS);

    // ---- text cross attention ----
    add_cvt_kernel<<<(nX + 255) / 256, 256, 0, stream>>>(X, QP, XQ16, nX);
    cvt(ct_w + (size_t)i * 768 * 256, W16, 768 * 256, stream);
    gemm(XQ16, W16,             ct_b + (size_t)i * 768,       Qb, MROWS,      256, 256, 0, 0, stream);
    gemm(MT16, W16 + 256 * 256, ct_b + (size_t)i * 768 + 256, Kb, BS * NTEXT, 256, 256, 0, 0, stream);
    gemm(MT16, W16 + 512 * 256, ct_b + (size_t)i * 768 + 512, Vb, BS * NTEXT, 256, 256, 0, 0, stream);
    attn_kernel<<<(BS * 8 * NQ + 7) / 8, 256, 0, stream>>>(
        Qb, Kb, Vb, tam, AO16, BS, 8, NQ, NTEXT, scale);
    cvt(ct_ow + (size_t)i * 65536, W16, 65536, stream);
    gemm(AO16, W16, ct_ob + (size_t)i * 256, T2, MROWS, 256, 256, 0, 0, stream);
    add_ln_kernel<<<(MROWS + 7) / 8, 256, 0, stream>>>(
        X, T2, cn_g + (size_t)i * 256, cn_b + (size_t)i * 256, X, nullptr, MROWS);

    // ---- multi-scale deformable attention ----
    add_cvt_kernel<<<(nX + 255) / 256, 256, 0, stream>>>(X, QP, XQ16, nX);
    cvt(off_w + (size_t)i * 65536, W16, 65536, stream);
    gemm(XQ16, W16, off_b + (size_t)i * 256, OFb, MROWS, 256, 256, 0, 0, stream);
    cvt(aw_w + (size_t)i * 32768, W16, 32768, stream);
    gemm(XQ16, W16, aw_b + (size_t)i * 128, AWb, MROWS, 128, 256, 0, 0, stream);
    softmax16_kernel<<<(MROWS * 8 + 255) / 256, 256, 0, stream>>>(AWb);
    cvt(vp_w + (size_t)i * 65536, W16, 65536, stream);
    gemm(MB16, W16, vp_b + (size_t)i * 256, VP16, BS * HWTOT, 256, 256, 0, 1, stream);
    deform_kernel<<<(MROWS * 8 + 7) / 8, 256, 0, stream>>>(OFb, AWb, RS, vr, VP16, AO16);
    cvt(op_w + (size_t)i * 65536, W16, 65536, stream);
    gemm(AO16, W16, op_b + (size_t)i * 256, T2, MROWS, 256, 256, 0, 0, stream);
    add_ln_kernel<<<(MROWS + 7) / 8, 256, 0, stream>>>(
        X, T2, n1_g + (size_t)i * 256, n1_b + (size_t)i * 256, X, X16, MROWS);

    // ---- FFN ----
    cvt(l1_w + (size_t)i * 2048 * 256, W16, 2048 * 256, stream);
    gemm(X16, W16, l1_b + (size_t)i * 2048, FF16, MROWS, 2048, 256, 1, 1, stream);
    cvt(l2_w + (size_t)i * 256 * 2048, W16, 256 * 2048, stream);
    gemm(FF16, W16, l2_b + (size_t)i * 256, T2, MROWS, 256, 2048, 0, 0, stream);
    // X16 <- layer output: feeds next layer's self-attn V projection
    add_ln_kernel<<<(MROWS + 7) / 8, 256, 0, stream>>>(
        X, T2, n3_g + (size_t)i * 256, n3_b + (size_t)i * 256, X, X16, MROWS);

    // ---- per-layer output norm -> hs[i] ----
    add_ln_kernel<<<(MROWS + 7) / 8, 256, 0, stream>>>(
        X, nullptr, nm_g, nm_b, out + (size_t)i * MROWS * 256, nullptr, MROWS);
  }
}
